// SparseEncoderVoxelNeXtFuse_7370163880458
// MI455X (gfx1250) — compile-verified, matrix operand order checked by reading, not run on hardware
//
#include <hip/hip_runtime.h>
#include <hip/hip_bf16.h>
#include <stdint.h>

// CDNA5 (gfx1250) wave32 WMMA sparse-conv encoder.
// Core op: gathered GEMM  Y[N,Co] = sum_k sum_c X[nbr[n,k],c] * W[k,c,Co]
// executed as v_wmma_f32_16x16x32_f16 over K = 27*Ci. A panels are staged to
// LDS with global_load_async_to_lds_b128 (ASYNCcnt), double-buffered so the
// gather DMA overlaps the matrix math. B is pre-packed to fragment layout.

typedef __attribute__((ext_vector_type(16))) _Float16 v16h;
typedef __attribute__((ext_vector_type(8)))  float    v8f;

// ------------------------------------------------------------------ utility
__global__ void k_zero_f32(float* p, long n) {
  long i = (long)blockIdx.x * blockDim.x + threadIdx.x;
  if (i < n) p[i] = 0.f;
}

__global__ void k_pad_feats(const float* __restrict__ f, _Float16* __restrict__ o, int n0) {
  int i = blockIdx.x * blockDim.x + threadIdx.x;
  if (i >= n0 * 16) return;
  int r = i >> 4, c = i & 15;
  o[i] = (c < 5) ? (_Float16)f[r * 5 + c] : (_Float16)0.f;
}

__global__ void k_f32_to_f16(const float* __restrict__ x, _Float16* __restrict__ y, long n) {
  long i = (long)blockIdx.x * blockDim.x + threadIdx.x;
  if (i < n) y[i] = (_Float16)x[i];
}

__global__ void k_segsum(const _Float16* __restrict__ x, const int* __restrict__ seg,
                         float* __restrict__ xb, int n, int C) {
  long i = (long)blockIdx.x * blockDim.x + threadIdx.x;
  if (i >= (long)n * C) return;
  int r = (int)(i / C), c = (int)(i % C);
  atomicAdd(&xb[(long)seg[r] * C + c], (float)x[i]);
}

// -------------------------------------------------- weight repack (B frags)
// f32 W[Kn][Ci_w][Co] -> f16 Wp[KT][Co/16][32 lanes][16 f16]
// 16-bit B fragment (32x16, KxN): lane -> N = lane&15; VGPR v pair base
//   K = (v/4)*16 + (lane>>4)*8 + (v&3)*2.   K flattened as k*Ci + c.
__global__ void k_wprep(const float* __restrict__ W, _Float16* __restrict__ Wp,
                        int Kn, int Ci, int Ci_w, int Co, int KT) {
  long t = (long)blockIdx.x * blockDim.x + threadIdx.x;
  long total = (long)KT * 32 * Co;
  if (t >= total) return;
  int e    = (int)(t & 15);
  int lane = (int)((t >> 4) & 31);
  int nct  = Co >> 4;
  int ct   = (int)((t >> 9) % nct);
  int kt   = (int)(t / ((long)512 * nct));
  int p = e & 1, v = e >> 1;
  int kloc = (v >> 2) * 16 + (lane >> 4) * 8 + (v & 3) * 2 + p;
  int kg = kt * 32 + kloc;
  int d  = ct * 16 + (lane & 15);
  int k  = kg / Ci, c = kg - k * Ci;
  float val = (k < Kn && c < Ci_w) ? W[((long)k * Ci_w + c) * Co + d] : 0.f;
  Wp[t] = (_Float16)val;
}

// ------------------------------------------------------ gathered WMMA GEMM
// Block = 256 threads = 8 waves; block handles 16 output voxels; wave wv
// produces the 16x16 tile for output-channel columns [wv*16, wv*16+16).
// Per round: one 16x128-f16 gathered A panel group (4 K-steps) is DMA'd to
// LDS (one async b128 per thread, exact fit) while the previous panel group
// is consumed by 4 unrolled v_wmma_f32_16x16x32_f16 steps.
__global__ __launch_bounds__(256)
void k_sconv_wmma(const _Float16* __restrict__ X, const _Float16* __restrict__ Wp,
                  const int* __restrict__ nbr, float* __restrict__ Y,
                  const _Float16* __restrict__ Zp,
                  int Nout, int Ci, int Co, int Kn, int KT) {
  // Sole LDS object -> LDS offset 0; async-to-LDS VDST VGPR carries the raw
  // LDS byte address. The array address is passed into the asm as a dummy
  // operand so the compiler knows the asm writes it (keeps the ds_loads).
  __shared__ _Float16 At[2][16 * 128];
  int tid  = threadIdx.x;
  int lane = tid & 31;
  int wv   = tid >> 5;
  int tile = blockIdx.x * 16;
  int nct  = Co >> 4;

  int m   = tid >> 4;      // staging row 0..15
  int j   = tid & 15;      // 8-f16 chunk within the 128-wide panel group
  int vox = tile + m;
  int rounds = (KT + 3) >> 2;

  uint64_t at_escape = (uint64_t)(uintptr_t)&At[0][0];

  auto issue = [&](int r) {
    int flatK = r * 128 + j * 8;        // flattened K = k*Ci + c
    int k = flatK / Ci;
    const _Float16* src = Zp;           // zero page: OOB rows / empty nbrs / K tail
    if (vox < Nout && k < Kn) {
      int c   = flatK - k * Ci;
      int idx = nbr[(long)vox * Kn + k];
      if (idx >= 0) src = X + (long)idx * Ci + c;
    }
    uint32_t dst = (uint32_t)(((r & 1) << 12) + (m * 128 + j * 8) * 2); // LDS bytes
    uint64_t sa  = (uint64_t)(uintptr_t)src;
    asm volatile("global_load_async_to_lds_b128 %0, %1, off"
                 :: "v"(dst), "v"(sa), "s"(at_escape) : "memory");
  };

  v8f acc = {};
  int mrow = lane & 15, half = lane >> 4;

  auto wstep = [&](const uint32_t* Arow, int kt, int p) {
    union { v16h v; uint32_t u[8]; } a;
#pragma unroll
    for (int vv = 0; vv < 8; ++vv) {
      // 16-bit A fragment (16x32): K pair base per ISA layout
      int kk = (vv >> 2) * 16 + half * 8 + (vv & 3) * 2;
      a.u[vv] = Arow[(p * 32 + kk) >> 1];
    }
    v16h b = *(const v16h*)(Wp + ((((long)kt * nct + wv) * 32 + lane) << 4));
    acc = __builtin_amdgcn_wmma_f32_16x16x32_f16(false, a.v, false, b,
                                                 (short)0, acc, false, false);
  };

  issue(0);                                 // prologue: stage panel group 0

  for (int r = 0; r < rounds; ++r) {
    if (r + 1 < rounds) {
      issue(r + 1);                         // overlap next DMA with this math
      asm volatile("s_wait_asynccnt 0x1" ::: "memory");  // panel r landed
    } else {
      asm volatile("s_wait_asynccnt 0x0" ::: "memory");
    }
    __syncthreads();

    if (wv < nct) {
      const uint32_t* Arow = (const uint32_t*)&At[r & 1][mrow * 128];
      int steps = KT - r * 4;
      if (steps >= 4) {
#pragma unroll
        for (int p = 0; p < 4; ++p) wstep(Arow, r * 4 + p, p);
      } else {
        for (int p = 0; p < steps; ++p) wstep(Arow, r * 4 + p, p);
      }
    }
    __syncthreads();                        // buffer free before reuse (r+2)
  }

  // ---- write D: VGPR r -> M = r + (lane>=16)*8, N = lane&15 ----
  if (wv < nct) {
    int col = (wv << 4) + (lane & 15);
    int mb  = (lane >> 4) * 8;
#pragma unroll
    for (int rr = 0; rr < 8; ++rr) {
      int row = tile + mb + rr;
      if (row < Nout) Y[(long)row * Co + col] = acc[rr];
    }
  }
}

// ------------------------------------------------------------- batch norm
// Per-block LDS reduction (ds_add_f32) then one global atomic per channel.
__global__ __launch_bounds__(256)
void k_bnstat(const float* __restrict__ Y, float* __restrict__ st, long n, int C) {
  __shared__ float ls[256];                 // [C] sum, [C] sumsq (C <= 128)
  int t = threadIdx.x;
  if (t < 2 * C) ls[t] = 0.f;
  __syncthreads();
  long i = (long)blockIdx.x * blockDim.x + t;
  if (i < n) {
    int c = (int)(i % C);
    float y = Y[i];
    atomicAdd(&ls[c], y);
    atomicAdd(&ls[C + c], y * y);
  }
  __syncthreads();
  if (t < 2 * C) atomicAdd(&st[t], ls[t]);
}

__global__ void k_bnapply(const float* __restrict__ Y, const float* __restrict__ g,
                          const float* __restrict__ b, const float* __restrict__ st,
                          const _Float16* __restrict__ res, _Float16* oh, float* of,
                          long n, int C, float eps, float invN, int relu) {
  long i = (long)blockIdx.x * blockDim.x + threadIdx.x;
  if (i >= n) return;
  int c = (int)(i % C);
  float mean = st[c] * invN;
  float var  = st[C + c] * invN - mean * mean;
  float v = (Y[i] - mean) * rsqrtf(var + eps) * g[c] + b[c];
  if (res) v += (float)res[i];
  if (relu && v < 0.f) v = 0.f;
  if (oh) oh[i] = (_Float16)v;
  if (of) of[i] = v;
}

// ------------------------------------------------------------------- host
extern "C" void kernel_launch(void* const* d_in, const int* in_sizes, int n_in,
                              void* d_out, int out_size, void* d_ws, size_t ws_size,
                              hipStream_t stream) {
  // Input order: JAX pytree flatten with alphabetical dict keys.
  // top: bev_nbr, bev_seg, dn1..dn5, feats, nbr0..nbr5, params{...}
  enum {
    I_BEV_NBR = 0, I_BEV_SEG = 1,
    I_DN1 = 2, I_DN2, I_DN3, I_DN4, I_DN5,
    I_FEATS = 7,
    I_NBR0 = 8, I_NBR1, I_NBR2, I_NBR3, I_NBR4, I_NBR5,
    // params subtree (keys sorted): inp{b,g,w} out{b,g,w} s1{blk1,blk2}
    // s2..s6{bd,blk1,blk2,gd,wd} shared{b,bias,g,w}; blk = {b1,b2,g1,g2,w1,w2}
    P_INP_B = 14, P_INP_G, P_INP_W,
    P_OUT_B = 17, P_OUT_G, P_OUT_W,
    P_S1 = 20,                       // blk1: 20..25, blk2: 26..31
    P_S2 = 32, P_S3 = 47, P_S4 = 62, P_S5 = 77, P_S6 = 92, // bd, blk1(6), blk2(6), gd, wd
    P_SH_B = 107, P_SH_BIAS = 108, P_SH_G = 109, P_SH_W = 110
  };
  (void)n_in; (void)out_size; (void)ws_size;

  const int n0   = in_sizes[I_NBR0] / 27;
  const int n1   = in_sizes[I_DN1] / 27;
  const int n2   = in_sizes[I_DN2] / 27;
  const int n3   = in_sizes[I_DN3] / 27;
  const int n4   = in_sizes[I_DN4] / 27;
  const int n5   = in_sizes[I_DN5] / 27;
  const int nbev = in_sizes[I_BEV_NBR] / 9;

  // ---- workspace carve ----
  char* base = (char*)d_ws;
  size_t off = 0;
  auto carve = [&](size_t bytes) -> void* {
    void* p = base + off;
    off = (off + bytes + 255) & ~(size_t)255;
    return p;
  };
  _Float16* zp  = (_Float16*)carve(4096);                      // zero page
  float*    st  = (float*)carve(2 * 128 * sizeof(float));      // bn sum/sumsq
  _Float16* Wp  = (_Float16*)carve((size_t)108 * 32 * 128 * 2);// packed weights
  size_t nm = (size_t)n0;
  _Float16* bufX = (_Float16*)carve(nm * 128 * 2);
  _Float16* bufT = (_Float16*)carve(nm * 128 * 2);
  float*    Y    = (float*)carve(nm * 128 * 4);
  float*    xb   = (float*)carve((size_t)nbev * 128 * 4);

  auto gz = [](long n, int b) { return (unsigned)((n + b - 1) / b); };

  k_zero_f32<<<gz(1024, 256), 256, 0, stream>>>((float*)zp, 1024);

  auto conv = [&](const _Float16* Xf, const int* nbr, int Nout, int Ci, int Ciw,
                  int Kn, int Co, const float* W32) {
    int KT = (Kn * Ci + 31) / 32;
    long wtot = (long)KT * 32 * Co;
    k_wprep<<<gz(wtot, 256), 256, 0, stream>>>(W32, Wp, Kn, Ci, Ciw, Co, KT);
    k_sconv_wmma<<<gz(Nout, 16), 256, 0, stream>>>(Xf, Wp, nbr, Y, zp, Nout, Ci, Co, Kn, KT);
  };
  auto bnorm = [&](const float* g, const float* b, int N, int C, float eps, int relu,
                   const _Float16* res, _Float16* oh, float* of) {
    k_zero_f32<<<1, 256, 0, stream>>>(st, 256);
    long n = (long)N * C;
    k_bnstat<<<gz(n, 256), 256, 0, stream>>>(Y, st, n, C);
    k_bnapply<<<gz(n, 256), 256, 0, stream>>>(Y, g, b, st, res, oh, of, n, C, eps, 1.0f / N, relu);
  };
  auto blockf = [&](const int* nbr, int N, int C, int pb) {
    const float* b1 = (const float*)d_in[pb + 0];
    const float* b2 = (const float*)d_in[pb + 1];
    const float* g1 = (const float*)d_in[pb + 2];
    const float* g2 = (const float*)d_in[pb + 3];
    const float* w1 = (const float*)d_in[pb + 4];
    const float* w2 = (const float*)d_in[pb + 5];
    conv(bufX, nbr, N, C, C, 27, C, w1);
    bnorm(g1, b1, N, C, 1e-3f, 1, nullptr, bufT, nullptr);
    conv(bufT, nbr, N, C, C, 27, C, w2);
    bnorm(g2, b2, N, C, 1e-3f, 1, bufX, bufX, nullptr);   // relu(x + bn(conv))
  };
  auto stage = [&](int pbase, const int* dn, const int* nbr, int Nout, int Ci, int Co) {
    const float* bd = (const float*)d_in[pbase];
    const float* gd = (const float*)d_in[pbase + 13];
    const float* wd = (const float*)d_in[pbase + 14];
    conv(bufX, dn, Nout, Ci, Ci, 27, Co, wd);
    bnorm(gd, bd, Nout, Co, 1e-3f, 1, nullptr, bufT, nullptr);
    { _Float16* t = bufX; bufX = bufT; bufT = t; }
    blockf(nbr, Nout, Co, pbase + 1);
    blockf(nbr, Nout, Co, pbase + 7);
  };

  // ---- input conv (Ci 5 padded to 16) ----
  k_pad_feats<<<gz((long)n0 * 16, 256), 256, 0, stream>>>((const float*)d_in[I_FEATS], bufX, n0);
  conv(bufX, (const int*)d_in[I_NBR0], n0, 16, 5, 27, 16, (const float*)d_in[P_INP_W]);
  bnorm((const float*)d_in[P_INP_G], (const float*)d_in[P_INP_B], n0, 16, 1e-3f, 1,
        nullptr, bufT, nullptr);
  { _Float16* t = bufX; bufX = bufT; bufT = t; }

  blockf((const int*)d_in[I_NBR0], n0, 16, P_S1 + 0);
  blockf((const int*)d_in[I_NBR0], n0, 16, P_S1 + 6);

  stage(P_S2, (const int*)d_in[I_DN1], (const int*)d_in[I_NBR1], n1, 16, 32);
  stage(P_S3, (const int*)d_in[I_DN2], (const int*)d_in[I_NBR2], n2, 32, 64);
  stage(P_S4, (const int*)d_in[I_DN3], (const int*)d_in[I_NBR3], n3, 64, 128);
  stage(P_S5, (const int*)d_in[I_DN4], (const int*)d_in[I_NBR4], n4, 128, 128);
  stage(P_S6, (const int*)d_in[I_DN5], (const int*)d_in[I_NBR5], n5, 128, 128);

  // ---- BEV: scatter-add then two 3x3 (K=9) convs ----
  k_zero_f32<<<gz((long)nbev * 128, 256), 256, 0, stream>>>(xb, (long)nbev * 128);
  k_segsum<<<gz((long)n5 * 128, 256), 256, 0, stream>>>(bufX, (const int*)d_in[I_BEV_SEG], xb, n5, 128);
  k_f32_to_f16<<<gz((long)nbev * 128, 256), 256, 0, stream>>>(xb, bufT, (long)nbev * 128);
  { _Float16* t = bufX; bufX = bufT; bufT = t; }

  conv(bufX, (const int*)d_in[I_BEV_NBR], nbev, 128, 128, 9, 128, (const float*)d_in[P_OUT_W]);
  bnorm((const float*)d_in[P_OUT_G], (const float*)d_in[P_OUT_B], nbev, 128, 1e-3f, 1,
        nullptr, bufT, nullptr);
  { _Float16* t = bufX; bufX = bufT; bufT = t; }

  // shared conv; its bias shifts the mean only, so it cancels inside the BN
  conv(bufX, (const int*)d_in[I_BEV_NBR], nbev, 128, 128, 9, 128, (const float*)d_in[P_SH_W]);
  bnorm((const float*)d_in[P_SH_G], (const float*)d_in[P_SH_B], nbev, 128, 1e-5f, 1,
        nullptr, nullptr, (float*)d_out);
}
